// SA_9998683865123
// MI455X (gfx1250) — compile-verified, matrix operand order checked by reading, not run on hardware
//
#include <hip/hip_runtime.h>

// ---------------------------------------------------------------------------
// Sigmoid self-attention-like op on MI455X (gfx1250), bf16 WMMA pipeline.
//   q = Wq@x+bq ; k = Wk@x+bk ; v = Wv@x+bv
//   s = q^T k ; alpha = sigmoid(s) ; o = v @ alpha
// Every GEMM: async global->LDS double-buffered staging (ASYNCcnt) feeding
// v_wmma_f32_16x16x32_bf16 from ds_load_b128 fragments.
// ---------------------------------------------------------------------------

typedef __bf16 bf16_t;
typedef __attribute__((ext_vector_type(8)))  __bf16 v8bf;
typedef __attribute__((ext_vector_type(16))) __bf16 v16bf;
typedef __attribute__((ext_vector_type(8)))  float  v8f;
typedef __attribute__((ext_vector_type(4)))  int    v4i;

typedef __attribute__((address_space(1))) v4i gv4i;   // global
typedef __attribute__((address_space(3))) v4i lv4i;   // LDS

#define B_  4
#define C_  512
#define N_  4096
#define O_  1000
#define OP_ 1024      // O padded to multiple of 32 (zero rows)
#define LPAD 40       // LDS row pitch in bf16 (80B): conflict-free frag reads

// --- async 16B global->LDS copy (ASYNCcnt) ---------------------------------
__device__ __forceinline__ void sa_cp_async16(const bf16_t* __restrict__ g,
                                              bf16_t* l) {
#if __has_builtin(__builtin_amdgcn_global_load_async_to_lds_b128)
  __builtin_amdgcn_global_load_async_to_lds_b128(
      (gv4i*)(v4i*)g, (lv4i*)(v4i*)l, 0, 0);
#else
  *(v8bf*)l = *(const v8bf*)g;   // synchronous fallback
#endif
}

__device__ __forceinline__ void sa_wait_async() {
#if __has_builtin(__builtin_amdgcn_s_wait_asynccnt)
  __builtin_amdgcn_s_wait_asynccnt(0);
#else
  asm volatile("s_wait_asynccnt 0" ::: "memory");
#endif
}

// Stage a (rows x 32) bf16 tile at column k0 of row-major [*][ldg] into LDS.
// 256 threads; each segment is 8 bf16 (16B).
__device__ __forceinline__ void sa_stage_tile(const bf16_t* __restrict__ g,
                                              int ldg, int k0, bf16_t* lds,
                                              int rows, int tid) {
  int segs = rows << 2;                       // 4 segments of 8 per row
  for (int s = tid; s < segs; s += 256) {
    int r = s >> 2, c = (s & 3) << 3;
    sa_cp_async16(g + (size_t)r * ldg + k0 + c, lds + r * LPAD + c);
  }
}

// --- fragment loader (LDS) --------------------------------------------------
// A-frag 16x32 (MxK) and B-frag 32x16 (KxN) share the lane pattern:
// lane l -> row (l&15), K-base ((l>>4)*8); 8 bf16 at K, 8 bf16 at K+16.
__device__ __forceinline__ v16bf sa_frag_merge(v8bf lo, v8bf hi) {
  v16bf r;
#pragma unroll
  for (int i = 0; i < 8; ++i) { r[i] = lo[i]; r[i + 8] = hi[i]; }
  return r;
}

__device__ __forceinline__ v16bf sa_load_frag_l(const bf16_t* tile, int row0) {
  int lane = threadIdx.x & 31;
  const bf16_t* p = tile + (row0 + (lane & 15)) * LPAD + ((lane >> 4) << 3);
  return sa_frag_merge(*(const v8bf*)p, *(const v8bf*)(p + 16));
}

__device__ __forceinline__ v8f sa_wmma(v16bf a, v16bf b, v8f c) {
  return __builtin_amdgcn_wmma_f32_16x16x32_bf16(false, a, false, b,
                                                 (short)0, c, false, false);
}

__device__ __forceinline__ float sa_sigmoid(float x) {
  return __builtin_amdgcn_rcpf(1.0f + __builtin_amdgcn_exp2f(x * -1.44269504f));
}

// ---------------------------------------------------------------------------
// LDS-staged 128x256 GEMM core: A rows at Ab (ld), B rows at Bb (ld), K=Kdim.
// Double-buffered async staging; 8 waves 2(M) x 4(N), 64x64 per wave.
// One barrier per k-step: wait own asynccnt -> barrier (tile kt visible,
// everyone done reading kt-1) -> issue fills for kt+1 -> compute kt.
// ---------------------------------------------------------------------------
__device__ __forceinline__ void sa_gemm_core(const bf16_t* __restrict__ Ab, int lda,
                                             const bf16_t* __restrict__ Bb, int ldb,
                                             int Kdim, bf16_t* lA, bf16_t* lB,
                                             int wm, int wn, v8f acc[4][4]) {
  const int tid = threadIdx.x;
  const int szA = 128 * LPAD, szB = 256 * LPAD;
  sa_stage_tile(Ab, lda, 0, lA, 128, tid);
  sa_stage_tile(Bb, ldb, 0, lB, 256, tid);
  const int steps = Kdim >> 5;
  for (int kt = 0; kt < steps; ++kt) {
    int cur = kt & 1;
    sa_wait_async();
    __syncthreads();
    if (kt + 1 < steps) {
      sa_stage_tile(Ab, lda, (kt + 1) << 5, lA + (cur ^ 1) * szA, 128, tid);
      sa_stage_tile(Bb, ldb, (kt + 1) << 5, lB + (cur ^ 1) * szB, 256, tid);
    }
    const bf16_t* tA = lA + cur * szA + (wm * 64) * LPAD;
    const bf16_t* tB = lB + cur * szB + (wn * 64) * LPAD;
    v16bf af[4], bfr[4];
#pragma unroll
    for (int i = 0; i < 4; ++i) af[i]  = sa_load_frag_l(tA, 16 * i);
#pragma unroll
    for (int j = 0; j < 4; ++j) bfr[j] = sa_load_frag_l(tB, 16 * j);
#pragma unroll
    for (int i = 0; i < 4; ++i)
#pragma unroll
      for (int j = 0; j < 4; ++j)
        acc[i][j] = sa_wmma(af[i], bfr[j], acc[i][j]);
  }
}

// ---------------------------------------------------------------------------
// K0a: LDS-tiled transpose+convert: x[b][c][n] f32 -> Xt[b][n][c] bf16
// ---------------------------------------------------------------------------
__global__ void SA_pack_x(const float* __restrict__ x, bf16_t* __restrict__ Xt) {
  __shared__ bf16_t tile[32][33];
  int b  = blockIdx.z;
  int n0 = blockIdx.x * 32;
  int c0 = blockIdx.y * 32;
  int tx = threadIdx.x, ty = threadIdx.y;
  const float* xb = x + (size_t)b * C_ * N_;
#pragma unroll
  for (int i = ty; i < 32; i += 8)
    tile[i][tx] = (bf16_t)xb[(size_t)(c0 + i) * N_ + (n0 + tx)];
  __syncthreads();
  bf16_t* ob = Xt + (size_t)b * N_ * C_;
#pragma unroll
  for (int i = ty; i < 32; i += 8)
    ob[(size_t)(n0 + i) * C_ + (c0 + tx)] = tile[tx][i];
}

// ---------------------------------------------------------------------------
// K0b: pack weights to bf16, pad O->OP with zeros, pad biases.
// ---------------------------------------------------------------------------
__global__ void SA_pack_w(const float* __restrict__ Wq, const float* __restrict__ bq,
                          const float* __restrict__ Wk, const float* __restrict__ bk,
                          const float* __restrict__ Wv, const float* __restrict__ bv,
                          bf16_t* __restrict__ Wqb, bf16_t* __restrict__ Wkb,
                          bf16_t* __restrict__ Wvb,
                          float* __restrict__ bqb, float* __restrict__ bkb,
                          float* __restrict__ bvb) {
  int idx = blockIdx.x * blockDim.x + threadIdx.x;
  if (idx < OP_ * C_) {
    int o = idx / C_, c = idx % C_;
    Wqb[idx] = (o < O_) ? (bf16_t)Wq[(size_t)o * C_ + c] : (bf16_t)0.0f;
    Wkb[idx] = (o < O_) ? (bf16_t)Wk[(size_t)o * C_ + c] : (bf16_t)0.0f;
  }
  if (idx < C_ * C_) Wvb[idx] = (bf16_t)Wv[idx];
  if (idx < OP_)     { bqb[idx] = (idx < O_) ? bq[idx] : 0.0f;
                       bkb[idx] = (idx < O_) ? bk[idx] : 0.0f; }
  if (idx < C_)      bvb[idx] = bv[idx];
}

// ---------------------------------------------------------------------------
// K1: projection GEMM  D[M][Ncols] = A[M][K] * B[Ncols][K]^T + bias
//   LDS-staged core; biasOnN=1 -> bias by column (q,k), 0 -> by row (v).
// ---------------------------------------------------------------------------
__global__ void __launch_bounds__(256)
SA_proj_gemm(const bf16_t* __restrict__ Abase, long aStride, int lda,
             const bf16_t* __restrict__ Bbase, long bStride, int ldb,
             bf16_t* __restrict__ Cbase, long cStride, int ldc,
             const float* __restrict__ bias, int biasOnN, int Kdim) {
  __shared__ __align__(16) bf16_t lA[2 * 128 * LPAD];
  __shared__ __align__(16) bf16_t lB[2 * 256 * LPAD];
  int b = blockIdx.z;
  const bf16_t* A  = Abase + (size_t)b * aStride + (size_t)(blockIdx.x * 128) * lda;
  const bf16_t* Bm = Bbase + (size_t)b * bStride + (size_t)(blockIdx.y * 256) * ldb;
  bf16_t*       Cm = Cbase + (size_t)b * cStride;

  int wave = threadIdx.x >> 5, lane = threadIdx.x & 31;
  int wm = wave >> 2, wn = wave & 3;

  v8f acc[4][4] = {};
  sa_gemm_core(A, lda, Bm, ldb, Kdim, lA, lB, wm, wn, acc);

  int M0 = blockIdx.x * 128 + wm * 64;
  int N0 = blockIdx.y * 256 + wn * 64;
  int ln = lane & 15, lh = lane >> 4;
#pragma unroll
  for (int i = 0; i < 4; ++i)
#pragma unroll
    for (int j = 0; j < 4; ++j) {
      int ncol = N0 + 16 * j + ln;
      float bN = biasOnN ? bias[ncol] : 0.0f;
#pragma unroll
      for (int r = 0; r < 8; ++r) {
        int mrow = M0 + 16 * i + r + 8 * lh;
        float v = acc[i][j][r] + (biasOnN ? bN : bias[mrow]);
        Cm[(size_t)mrow * ldc + ncol] = (bf16_t)v;
      }
    }
}

// ---------------------------------------------------------------------------
// K2: s = Qt * Kt^T (K=OP_), fused sigmoid, store alpha^T[b][m][n] bf16.
//   Transposed store is one contiguous 128-bit store per fragment.
// ---------------------------------------------------------------------------
__global__ void __launch_bounds__(256)
SA_s_sigmoid(const bf16_t* __restrict__ Qt, const bf16_t* __restrict__ Kt,
             bf16_t* __restrict__ AlphaT) {
  __shared__ __align__(16) bf16_t lA[2 * 128 * LPAD];
  __shared__ __align__(16) bf16_t lB[2 * 256 * LPAD];
  int b = blockIdx.z;
  const bf16_t* Qb = Qt + (size_t)b * N_ * OP_ + (size_t)(blockIdx.x * 128) * OP_;
  const bf16_t* Kb = Kt + (size_t)b * N_ * OP_ + (size_t)(blockIdx.y * 256) * OP_;
  bf16_t*       Ab = AlphaT + (size_t)b * (size_t)N_ * N_;

  int wave = threadIdx.x >> 5, lane = threadIdx.x & 31;
  int wm = wave >> 2, wn = wave & 3;

  v8f acc[4][4] = {};
  sa_gemm_core(Qb, OP_, Kb, OP_, OP_, lA, lB, wm, wn, acc);

  int M0 = blockIdx.x * 128 + wm * 64;   // n rows
  int N0 = blockIdx.y * 256 + wn * 64;   // m cols
  int ln = lane & 15, lh = lane >> 4;
#pragma unroll
  for (int i = 0; i < 4; ++i)
#pragma unroll
    for (int j = 0; j < 4; ++j) {
      int m    = N0 + 16 * j + ln;        // fixed per lane
      int nrow = M0 + 16 * i + 8 * lh;    // 8 contiguous n per lane
      v8bf outv;
#pragma unroll
      for (int r = 0; r < 8; ++r) outv[r] = (bf16_t)sa_sigmoid(acc[i][j][r]);
      *(v8bf*)(Ab + (size_t)m * N_ + nrow) = outv;
    }
}

// ---------------------------------------------------------------------------
// K3: o[c][m] = V[c][:] * alpha[:, m]  (K = N_), f32 output (coalesced b32).
// ---------------------------------------------------------------------------
__global__ void __launch_bounds__(256)
SA_o_gemm(const bf16_t* __restrict__ Vb, const bf16_t* __restrict__ AlphaT,
          float* __restrict__ Out) {
  __shared__ __align__(16) bf16_t lA[2 * 128 * LPAD];
  __shared__ __align__(16) bf16_t lB[2 * 256 * LPAD];
  int b = blockIdx.z;
  const bf16_t* Vm = Vb + (size_t)b * C_ * N_ + (size_t)(blockIdx.x * 128) * N_;
  const bf16_t* Am = AlphaT + (size_t)b * (size_t)N_ * N_
                   + (size_t)(blockIdx.y * 256) * N_;
  float* Om = Out + (size_t)b * C_ * N_;

  int wave = threadIdx.x >> 5, lane = threadIdx.x & 31;
  int wm = wave >> 2, wn = wave & 3;

  v8f acc[4][4] = {};
  sa_gemm_core(Vm, N_, Am, N_, N_, lA, lB, wm, wn, acc);

  int M0 = blockIdx.x * 128 + wm * 64;   // c rows
  int N0 = blockIdx.y * 256 + wn * 64;   // m cols
  int ln = lane & 15, lh = lane >> 4;
#pragma unroll
  for (int i = 0; i < 4; ++i)
#pragma unroll
    for (int j = 0; j < 4; ++j) {
      int ncol = N0 + 16 * j + ln;
#pragma unroll
      for (int r = 0; r < 8; ++r) {
        int crow = M0 + 16 * i + r + 8 * lh;
        Om[(size_t)crow * N_ + ncol] = acc[i][j][r];
      }
    }
}

// ---------------------------------------------------------------------------
extern "C" void kernel_launch(void* const* d_in, const int* in_sizes, int n_in,
                              void* d_out, int out_size, void* d_ws, size_t ws_size,
                              hipStream_t stream) {
  const float* x  = (const float*)d_in[0];
  const float* Wq = (const float*)d_in[1];
  const float* bq = (const float*)d_in[2];
  const float* Wk = (const float*)d_in[3];
  const float* bk = (const float*)d_in[4];
  const float* Wv = (const float*)d_in[5];
  const float* bv = (const float*)d_in[6];
  float* out = (float*)d_out;

  char* ws = (char*)d_ws;
  size_t off = 0;
  auto take = [&](size_t bytes) -> char* {
    char* p = ws + off;
    off = (off + bytes + 255) & ~(size_t)255;
    return p;
  };

  bf16_t* Xt   = (bf16_t*)take((size_t)B_ * N_ * C_ * 2);
  bf16_t* Wqb  = (bf16_t*)take((size_t)OP_ * C_ * 2);
  bf16_t* Wkb  = (bf16_t*)take((size_t)OP_ * C_ * 2);
  bf16_t* Wvb  = (bf16_t*)take((size_t)C_ * C_ * 2);
  float*  bqb  = (float*) take((size_t)OP_ * 4);
  float*  bkb  = (float*) take((size_t)OP_ * 4);
  float*  bvb  = (float*) take((size_t)C_ * 4);
  bf16_t* Qt   = (bf16_t*)take((size_t)B_ * N_ * OP_ * 2);  // [b][n][o]
  bf16_t* Kt   = (bf16_t*)take((size_t)B_ * N_ * OP_ * 2);  // [b][m][o]
  bf16_t* Vbuf = (bf16_t*)take((size_t)B_ * C_ * N_ * 2);   // [b][c][n]
  bf16_t* AlT  = (bf16_t*)take((size_t)B_ * (size_t)N_ * N_ * 2); // [b][m][n]
  (void)ws_size; (void)in_sizes; (void)n_in; (void)out_size;

  SA_pack_x<<<dim3(N_ / 32, C_ / 32, B_), dim3(32, 8), 0, stream>>>(x, Xt);
  SA_pack_w<<<dim3((OP_ * C_) / 256, 1, 1), dim3(256), 0, stream>>>(
      Wq, bq, Wk, bk, Wv, bv, Wqb, Wkb, Wvb, bqb, bkb, bvb);

  // Qt[b][n][o]: A = Xt[b] (M=4096), B = Wqb (Ncols=1024), bias on N
  SA_proj_gemm<<<dim3(N_ / 128, OP_ / 256, B_), dim3(256), 0, stream>>>(
      Xt, (long)N_ * C_, C_, Wqb, 0L, C_, Qt, (long)N_ * OP_, OP_, bqb, 1, C_);
  SA_proj_gemm<<<dim3(N_ / 128, OP_ / 256, B_), dim3(256), 0, stream>>>(
      Xt, (long)N_ * C_, C_, Wkb, 0L, C_, Kt, (long)N_ * OP_, OP_, bkb, 1, C_);
  // V[b][c][n]: A = Wvb (M=512), B = Xt[b] (Ncols=4096), bias on M
  SA_proj_gemm<<<dim3(C_ / 128, N_ / 256, B_), dim3(256), 0, stream>>>(
      Wvb, 0L, C_, Xt, (long)N_ * C_, C_, Vbuf, (long)C_ * N_, N_, bvb, 0, C_);

  SA_s_sigmoid<<<dim3(N_ / 128, N_ / 256, B_), dim3(256), 0, stream>>>(Qt, Kt, AlT);
  SA_o_gemm<<<dim3(C_ / 128, N_ / 256, B_), dim3(256), 0, stream>>>(Vbuf, AlT, out);
}
